// TopKSAE_1245540515905
// MI455X (gfx1250) — matrix operand order repaired and checked by reading, not compiled, not run on hardware
//
#include <hip/hip_runtime.h>
#include <hip/hip_bf16.h>
#include <stdint.h>

#define D_MODEL 1024
#define D_SAE   16384
#define BATCH   4096
#define TOPK    32

typedef __bf16 v8bf  __attribute__((ext_vector_type(8)));
typedef __bf16 v16bf __attribute__((ext_vector_type(16)));
typedef float  v8f   __attribute__((ext_vector_type(8)));

union ABFrag { v16bf v; v8bf h[2]; };

__device__ __forceinline__ unsigned short f2bf(float f) {
  unsigned u = __builtin_bit_cast(unsigned, f);
  u += 0x7FFFu + ((u >> 16) & 1u);          // round-to-nearest-even
  return (unsigned short)(u >> 16);
}
__device__ __forceinline__ unsigned pk2(float lo, float hi) {
  return (unsigned)f2bf(lo) | ((unsigned)f2bf(hi) << 16);
}

// ---------------- Kernel 1: encoder GEMM, bf16 WMMA, f32 accumulate ----------------
// Double-buffered LDS (one barrier per K-tile) + register prefetch of the next tile
// issued before the WMMA block so global latency is hidden behind matrix work.
#define BM 128
#define BN 128
#define BK 64
#define KTILES (D_MODEL / BK)   // 16
#define LSTR 80                 // LDS row stride in bf16 elems (BK + 16 pad -> 160B)

__global__ __launch_bounds__(256) void sae_encode_wmma(
    const float* __restrict__ x, const float* __restrict__ W_enc,
    const float* __restrict__ b_enc, const float* __restrict__ b_pre,
    float* __restrict__ lat) {
  __shared__ unsigned short As[2][BM * LSTR];   // x - b_pre, bf16, [m][k]
  __shared__ unsigned short Bs[2][BN * LSTR];   // W_enc tile, bf16, [n][k]

  const int tid  = threadIdx.x;
  const int lane = tid & 31;
  const int wave = tid >> 5;
  const int wm = wave & 3;      // 4 waves along M: 32 rows each
  const int wn = wave >> 2;     // 2 waves along N: 64 cols each
  const int ll = lane & 15;
  const int lh = lane >> 4;

  const int mBase = blockIdx.y * BM;
  const int nBase = blockIdx.x * BN;

  // A staging coords (float4 granularity): 128x64 f32 tile, 8 float4 per thread
  const int am0 = tid >> 4;           // +16 per it (idx = tid + it*256)
  const int akq = tid & 15;
  // B staging coords: n fixed per thread, k strided
  const int bn  = tid & 127;
  const int bk0 = (tid >> 7) * 4;

  // Fold encoder bias into the accumulators (C/D layout: N = lane%16, same per VGPR)
  float be[4];
  #pragma unroll
  for (int ni = 0; ni < 4; ++ni)
    be[ni] = b_enc[nBase + wn * 64 + ni * 16 + ll];

  v8f acc[2][4];
  #pragma unroll
  for (int mi = 0; mi < 2; ++mi)
    #pragma unroll
    for (int ni = 0; ni < 4; ++ni)
      #pragma unroll
      for (int r = 0; r < 8; ++r)
        acc[mi][ni][r] = be[ni];

  // Raw staged registers for the next K-tile
  float4 ar[8];
  float  br[8][4];

  // ---- raw global loads of one K-tile into registers ----
  auto load_tile = [&](int k0) {
    #pragma unroll
    for (int it = 0; it < 8; ++it) {
      int m = am0 + it * 16;
      ar[it] = *(const float4*)(x + (size_t)(mBase + m) * D_MODEL + k0 + akq * 4);
    }
    #pragma unroll
    for (int it = 0; it < 8; ++it) {
      int k = bk0 + it * 8;
      const float* wp = W_enc + (size_t)(k0 + k) * D_SAE + nBase + bn;
      br[it][0] = wp[0];
      br[it][1] = wp[(size_t)D_SAE];
      br[it][2] = wp[2 * (size_t)D_SAE];
      br[it][3] = wp[3 * (size_t)D_SAE];
    }
  };

  // ---- convert staged registers to bf16 and store into LDS buffer `buf` ----
  auto store_tile = [&](int buf, int k0) {
    #pragma unroll
    for (int it = 0; it < 8; ++it) {
      int m = am0 + it * 16;
      const float4 bp = *(const float4*)(b_pre + k0 + akq * 4);  // L0-hot broadcast
      uint2 p;
      p.x = pk2(ar[it].x - bp.x, ar[it].y - bp.y);
      p.y = pk2(ar[it].z - bp.z, ar[it].w - bp.w);
      *(uint2*)&As[buf][m * LSTR + akq * 4] = p;
    }
    #pragma unroll
    for (int it = 0; it < 8; ++it) {
      int k = bk0 + it * 8;
      uint2 p;
      p.x = pk2(br[it][0], br[it][1]);
      p.y = pk2(br[it][2], br[it][3]);
      *(uint2*)&Bs[buf][bn * LSTR + k] = p;
    }
  };

  // ---- 16 WMMAs on LDS buffer `buf` ----
  auto compute_tile = [&](int buf) {
    #pragma unroll
    for (int ks = 0; ks < 2; ++ks) {
      // A frag: lane ll = row; K elems lh*8+{0..7} and 16+lh*8+{0..7}
      ABFrag af[2];
      #pragma unroll
      for (int mi = 0; mi < 2; ++mi) {
        int base = (wm * 32 + mi * 16 + ll) * LSTR + ks * 32 + lh * 8;
        af[mi].h[0] = *(const v8bf*)&As[buf][base];
        af[mi].h[1] = *(const v8bf*)&As[buf][base + 16];
      }
      // B frag: lane ll = col; K elems lh*8+{0..7} and 16+lh*8+{0..7}
      ABFrag bfB[4];
      #pragma unroll
      for (int ni = 0; ni < 4; ++ni) {
        int base = (wn * 64 + ni * 16 + ll) * LSTR + ks * 32 + lh * 8;
        bfB[ni].h[0] = *(const v8bf*)&Bs[buf][base];
        bfB[ni].h[1] = *(const v8bf*)&Bs[buf][base + 16];
      }
      #pragma unroll
      for (int mi = 0; mi < 2; ++mi)
        #pragma unroll
        for (int ni = 0; ni < 4; ++ni)
          acc[mi][ni] = __builtin_amdgcn_wmma_f32_16x16x32_bf16(
              false, af[mi].v, false, bfB[ni].v, (short)0, acc[mi][ni], false, false);
    }
  };

  // Prologue: stage tile 0
  load_tile(0);
  store_tile(0, 0);
  __syncthreads();

  for (int t = 0; t < KTILES; ++t) {
    const int nxt = t + 1;
    if (nxt < KTILES) load_tile(nxt * BK);   // issue global loads first
    compute_tile(t & 1);                     // 16 WMMAs hide the load latency
    if (nxt < KTILES) store_tile(nxt & 1, nxt * BK);
    __syncthreads();                         // single barrier per K-tile
  }

  // Epilogue: C/D layout -> dense latents (row M = r + lh*8, col N = ll)
  #pragma unroll
  for (int mi = 0; mi < 2; ++mi)
    #pragma unroll
    for (int ni = 0; ni < 4; ++ni) {
      int m = mBase + wm * 32 + mi * 16 + lh * 8;
      int c = nBase + wn * 64 + ni * 16 + ll;
      float* p = lat + (size_t)m * D_SAE + c;
      #pragma unroll
      for (int r = 0; r < 8; ++r)
        p[(size_t)r * D_SAE] = acc[mi][ni][r];
    }
}

// ---------------- Kernel 2: per-row top-K, sparsify, decode ----------------
__global__ __launch_bounds__(256) void sae_topk_decode(
    const float* __restrict__ W_dec, const float* __restrict__ b_pre,
    float* __restrict__ xhat, float* __restrict__ lat,
    int* __restrict__ idx_out, float* __restrict__ val_out) {
  __shared__ float srow[D_SAE];    // 64KB
  __shared__ float sval[256];
  __shared__ int   sidx[256];
  __shared__ float topv[TOPK];
  __shared__ int   topi[TOPK];

  const int tid = threadIdx.x;
  const int row = blockIdx.x;
  float* latRow = lat + (size_t)row * D_SAE;

  // Async global->LDS copy of the dense latent row (CDNA5 async-to-LDS path)
  #pragma unroll
  for (int it = 0; it < 16; ++it) {
    int j4 = tid + it * 256;                              // float4 index
    unsigned lds = (unsigned)(uintptr_t)&srow[j4 * 4];    // low 32 bits = LDS offset
    unsigned long long g = (unsigned long long)(uintptr_t)(latRow + j4 * 4);
    asm volatile("global_load_async_to_lds_b128 %0, %1, off"
                 :: "v"(lds), "v"(g) : "memory");
  }
  asm volatile("s_wait_asynccnt 0" ::: "memory");
  __syncthreads();

  const float MARK = -3.402823466e38f;
  for (int it = 0; it < TOPK; ++it) {
    float best = MARK; int bidx = 0x7FFFFFFF;
    for (int j = tid; j < D_SAE; j += 256) {
      float v = srow[j];
      if (v > best) { best = v; bidx = j; }   // ascending j keeps lowest index on tie
    }
    sval[tid] = best; sidx[tid] = bidx;
    __syncthreads();
    #pragma unroll
    for (int s = 128; s > 0; s >>= 1) {
      if (tid < s) {
        float v2 = sval[tid + s]; int i2 = sidx[tid + s];
        float v1 = sval[tid];     int i1 = sidx[tid];
        if (v2 > v1 || (v2 == v1 && i2 < i1)) { sval[tid] = v2; sidx[tid] = i2; }
      }
      __syncthreads();
    }
    if (tid == 0) {
      topv[it] = sval[0]; topi[it] = sidx[0];
      srow[sidx[0]] = MARK;                   // remove winner
    }
    __syncthreads();
  }

  if (idx_out != nullptr && tid < TOPK) {
    idx_out[row * TOPK + tid] = topi[tid];
    val_out[row * TOPK + tid] = topv[tid];
  }

  // Rewrite the row as the sparse top-k scatter (zeros elsewhere)
  for (int j = tid; j < D_SAE; j += 256) {
    float o = 0.f;
    if (srow[j] == MARK) {
      #pragma unroll
      for (int t = 0; t < TOPK; ++t)
        if (topi[t] == j) o = topv[t];
    }
    latRow[j] = o;
  }

  // Decode: x_hat = b_pre + sum_k topv[k] * W_dec[topi[k], :]   (W_dec is L2-resident)
  int c0 = tid * 4;                // 256 threads * 4 = 1024 = D_MODEL
  float4 a = *(const float4*)(b_pre + c0);
  #pragma unroll
  for (int t = 0; t < TOPK; ++t) {
    float w = topv[t];
    const float4 wd = *(const float4*)(W_dec + (size_t)topi[t] * D_MODEL + c0);
    a.x += w * wd.x; a.y += w * wd.y; a.z += w * wd.z; a.w += w * wd.w;
  }
  *(float4*)(xhat + (size_t)row * D_MODEL + c0) = a;
}

// ---------------- Host launcher ----------------
extern "C" void kernel_launch(void* const* d_in, const int* in_sizes, int n_in,
                              void* d_out, int out_size, void* d_ws, size_t ws_size,
                              hipStream_t stream) {
  const float* x     = (const float*)d_in[0];
  const float* W_enc = (const float*)d_in[1];
  const float* b_enc = (const float*)d_in[2];
  const float* W_dec = (const float*)d_in[3];
  const float* b_pre = (const float*)d_in[4];

  float* out  = (float*)d_out;
  float* xhat = out;                                  // (B, d_model)
  float* lat  = out + (size_t)BATCH * D_MODEL;        // (B, d_sae): dense -> sparsified

  // Dict outputs (topk_indices, topk_values) follow in tree-flatten order if present.
  const long long full = (long long)BATCH * (D_MODEL + D_SAE + 2 * TOPK);
  int*   idx_out = nullptr;
  float* val_out = nullptr;
  if ((long long)out_size >= full) {
    float* tail = lat + (size_t)BATCH * D_SAE;
    idx_out = (int*)tail;
    val_out = tail + (size_t)BATCH * TOPK;
  } else if (ws_size >= (size_t)BATCH * TOPK * 8) {
    idx_out = (int*)d_ws;
    val_out = (float*)d_ws + (size_t)BATCH * TOPK;
  }

  dim3 gA(D_SAE / BN, BATCH / BM);  // (128, 32)
  sae_encode_wmma<<<gA, 256, 0, stream>>>(x, W_enc, b_enc, b_pre, lat);
  sae_topk_decode<<<BATCH, 256, 0, stream>>>(W_dec, b_pre, xhat, lat, idx_out, val_out);
}
